// StructuralPreservationLoss_67379446940398
// MI455X (gfx1250) — compile-verified, matrix operand order checked by reading, not run on hardware
//
#include <hip/hip_runtime.h>
#include <hip/hip_bf16.h>

typedef __attribute__((ext_vector_type(16))) _Float16 v16h;
typedef __attribute__((ext_vector_type(8)))  float    v8f;

#define NUM_NODES   50000
#define EMBED_DIM   128
#define NUM_POS     8192
#define NUM_NEG     (NUM_POS * 5)          // 40960
#define NUM_EDGES   (NUM_POS + NUM_NEG)    // 49152
#define MARGIN_F    0.1f
#define LAMBDA_F    1.0f

#define TILES            (NUM_EDGES / 16)          // 3072 wave-tiles
#define WAVES_PER_BLOCK  8
#define SIM_BLOCKS       (TILES / WAVES_PER_BLOCK) // 384

#define HINGE_THREADS    256
#define POS_PER_THREAD   4
#define POS_PER_BLOCK    (HINGE_THREADS * POS_PER_THREAD)  // 1024
#define POS_BLOCKS       (NUM_POS / POS_PER_BLOCK)         // 8
#define NEG_SLICES       32
#define NEG_SLICE        (NUM_NEG / NEG_SLICES)            // 1280 floats = 5 KB LDS
#define N_PARTIALS       (POS_BLOCKS * NEG_SLICES)         // 256

// ---------------------------------------------------------------------------
// Kernel A: per-edge dot products via v_wmma_f32_16x16x32_f16.
// One wave handles 16 edges: A rows = src embeddings, B cols = dst embeddings,
// accumulate over K=128 in 4 WMMA steps, then extract diag(D).
// ---------------------------------------------------------------------------
__global__ void sims_wmma_kernel(const float* __restrict__ emb,
                                 const long long* __restrict__ eidx,
                                 const long long* __restrict__ nidx,
                                 float* __restrict__ sims) {
    const int lane = threadIdx.x & 31;
    const int wave = threadIdx.x >> 5;
    const int tile = blockIdx.x * WAVES_PER_BLOCK + wave;   // 0..TILES-1 exact
    const int m    = lane & 15;        // edge within tile (matrix row/col)
    const int hlf  = lane >> 4;        // which K half this lane carries
    const int e    = tile * 16 + m;

    // Tile never straddles the pos/neg boundary (NUM_POS % 16 == 0): uniform.
    long long s, d;
    if (e < NUM_POS) {
        s = eidx[e];
        d = eidx[NUM_POS + e];
    } else {
        const int en = e - NUM_POS;
        s = nidx[en];
        d = nidx[NUM_NEG + en];
    }

    const float* sp = emb + (size_t)s * EMBED_DIM;
    const float* dp = emb + (size_t)d * EMBED_DIM;
    __builtin_prefetch(sp, 0, 0);      // global_prefetch_b8
    __builtin_prefetch(dp, 0, 0);

    v8f c = {};
    const int base = hlf * 8;          // lanes 0-15: K base 0; lanes 16-31: K base 8

#pragma unroll
    for (int k0 = 0; k0 < EMBED_DIM; k0 += 32) {
        v16h a, b;
        {
            const float* p = sp + k0 + base;
            float4 l0 = *(const float4*)(p);
            float4 l1 = *(const float4*)(p + 4);
            float4 h0 = *(const float4*)(p + 16);
            float4 h1 = *(const float4*)(p + 20);
            a[0]=(_Float16)l0.x; a[1]=(_Float16)l0.y; a[2]=(_Float16)l0.z; a[3]=(_Float16)l0.w;
            a[4]=(_Float16)l1.x; a[5]=(_Float16)l1.y; a[6]=(_Float16)l1.z; a[7]=(_Float16)l1.w;
            a[8]=(_Float16)h0.x; a[9]=(_Float16)h0.y; a[10]=(_Float16)h0.z; a[11]=(_Float16)h0.w;
            a[12]=(_Float16)h1.x; a[13]=(_Float16)h1.y; a[14]=(_Float16)h1.z; a[15]=(_Float16)h1.w;
        }
        {
            const float* p = dp + k0 + base;   // B: lane -> column N, mirrored K map
            float4 l0 = *(const float4*)(p);
            float4 l1 = *(const float4*)(p + 4);
            float4 h0 = *(const float4*)(p + 16);
            float4 h1 = *(const float4*)(p + 20);
            b[0]=(_Float16)l0.x; b[1]=(_Float16)l0.y; b[2]=(_Float16)l0.z; b[3]=(_Float16)l0.w;
            b[4]=(_Float16)l1.x; b[5]=(_Float16)l1.y; b[6]=(_Float16)l1.z; b[7]=(_Float16)l1.w;
            b[8]=(_Float16)h0.x; b[9]=(_Float16)h0.y; b[10]=(_Float16)h0.z; b[11]=(_Float16)h0.w;
            b[12]=(_Float16)h1.x; b[13]=(_Float16)h1.y; b[14]=(_Float16)h1.z; b[15]=(_Float16)h1.w;
        }
        c = __builtin_amdgcn_wmma_f32_16x16x32_f16(
                /*neg_a=*/false, a, /*neg_b=*/false, b,
                /*c_mod=*/(short)0, c, /*reuse_a=*/false, /*reuse_b=*/false);
    }

    // Diagonal extraction: D[r][r] is in VGPR r of lane r (M=0..7) and of
    // lane r+24 (M=8..15). sel = lane&7 picks the VGPR slot.
    const int sel = lane & 7;
    float diag = c[0];
    diag = (sel == 1) ? (float)c[1] : diag;
    diag = (sel == 2) ? (float)c[2] : diag;
    diag = (sel == 3) ? (float)c[3] : diag;
    diag = (sel == 4) ? (float)c[4] : diag;
    diag = (sel == 5) ? (float)c[5] : diag;
    diag = (sel == 6) ? (float)c[6] : diag;
    diag = (sel == 7) ? (float)c[7] : diag;

    if (lane < 8 || lane >= 24) {
        const int mm = (lane < 8) ? lane : (lane - 16);
        sims[tile * 16 + mm] = diag;
    }
}

// ---------------------------------------------------------------------------
// Kernel B: hinge partial sums over a (1024-pos x 1280-neg) sub-tile.
// Grid = (POS_BLOCKS, NEG_SLICES) = (8, 32). The neg slice (5 KB) is copied
// to LDS with CDNA5 async-to-LDS loads (ASYNCcnt), then each thread register-
// tiles 4 pos edges so one broadcast ds_load_b128 feeds 16 hinge evals.
// ---------------------------------------------------------------------------
__global__ void hinge_kernel(const float* __restrict__ pos,
                             const float* __restrict__ neg,
                             float* __restrict__ partials) {
    __shared__ float slds[NEG_SLICE];
    const int tid = threadIdx.x;

    // Async fill of the neg slice: global -> LDS without a VGPR round-trip.
    {
        const float* nslice = neg + (size_t)blockIdx.y * NEG_SLICE;
        const unsigned ldsbase = (unsigned)(unsigned long long)(const void*)&slds[0];
        for (int j = tid; j < NEG_SLICE / 4; j += HINGE_THREADS) {
            const unsigned goff  = (unsigned)j * 16u;          // byte offset (GVS)
            const unsigned laddr = ldsbase + (unsigned)j * 16u; // LDS byte addr
            asm volatile("global_load_async_to_lds_b128 %0, %1, %2"
                         :: "v"(laddr), "v"(goff), "s"(nslice)
                         : "memory");
        }
        asm volatile("s_wait_asynccnt 0x0" ::: "memory");
    }
    __syncthreads();

    // 4 pos edges per thread (contiguous float4, POS_PER_BLOCK-aligned base).
    const int i0 = blockIdx.x * POS_PER_BLOCK + tid * POS_PER_THREAD;
    const float4 p4 = *(const float4*)(pos + i0);
    const float c0 = MARGIN_F - p4.x;
    const float c1 = MARGIN_F - p4.y;
    const float c2 = MARGIN_F - p4.z;
    const float c3 = MARGIN_F - p4.w;

    float acc = 0.0f;
    const float4* slds4 = (const float4*)slds;
#pragma unroll 4
    for (int j = 0; j < NEG_SLICE / 4; ++j) {
        const float4 n = slds4[j];
        acc += fmaxf(c0 + n.x, 0.0f); acc += fmaxf(c0 + n.y, 0.0f);
        acc += fmaxf(c0 + n.z, 0.0f); acc += fmaxf(c0 + n.w, 0.0f);
        acc += fmaxf(c1 + n.x, 0.0f); acc += fmaxf(c1 + n.y, 0.0f);
        acc += fmaxf(c1 + n.z, 0.0f); acc += fmaxf(c1 + n.w, 0.0f);
        acc += fmaxf(c2 + n.x, 0.0f); acc += fmaxf(c2 + n.y, 0.0f);
        acc += fmaxf(c2 + n.z, 0.0f); acc += fmaxf(c2 + n.w, 0.0f);
        acc += fmaxf(c3 + n.x, 0.0f); acc += fmaxf(c3 + n.y, 0.0f);
        acc += fmaxf(c3 + n.z, 0.0f); acc += fmaxf(c3 + n.w, 0.0f);
    }

    // Deterministic block reduction (wave shuffle + LDS tree).
    for (int off = 16; off > 0; off >>= 1)
        acc += __shfl_down(acc, off, 32);
    __shared__ float red[HINGE_THREADS / 32];
    if ((tid & 31) == 0) red[tid >> 5] = acc;
    __syncthreads();
    if (tid < HINGE_THREADS / 32) {
        float v = red[tid];
        for (int off = (HINGE_THREADS / 64); off > 0; off >>= 1)
            v += __shfl_down(v, off, HINGE_THREADS / 32);
        if (tid == 0)
            partials[blockIdx.y * POS_BLOCKS + blockIdx.x] = v;
    }
}

// ---------------------------------------------------------------------------
// Kernel C: final deterministic reduction + scale.
// ---------------------------------------------------------------------------
__global__ void finalize_kernel(const float* __restrict__ partials,
                                float* __restrict__ out) {
    const int tid = threadIdx.x;
    float v = partials[tid];                 // exactly N_PARTIALS threads
    for (int off = 16; off > 0; off >>= 1)
        v += __shfl_down(v, off, 32);
    __shared__ float red[N_PARTIALS / 32];
    if ((tid & 31) == 0) red[tid >> 5] = v;
    __syncthreads();
    if (tid < N_PARTIALS / 32) {
        float t = red[tid];
        for (int off = (N_PARTIALS / 64); off > 0; off >>= 1)
            t += __shfl_down(t, off, N_PARTIALS / 32);
        if (tid == 0)
            out[0] = LAMBDA_F * (t / ((float)NUM_POS * (float)NUM_NEG));
    }
}

extern "C" void kernel_launch(void* const* d_in, const int* in_sizes, int n_in,
                              void* d_out, int out_size, void* d_ws, size_t ws_size,
                              hipStream_t stream) {
    const float*     emb  = (const float*)d_in[0];
    const long long* eidx = (const long long*)d_in[1];   // int64 [2, NUM_POS]
    const long long* nidx = (const long long*)d_in[2];   // int64 [2, NUM_NEG]

    float* sims     = (float*)d_ws;            // [NUM_EDGES]: pos then neg
    float* pos      = sims;                    // [NUM_POS]
    float* neg      = sims + NUM_POS;          // [NUM_NEG]
    float* partials = sims + NUM_EDGES;        // [N_PARTIALS]

    sims_wmma_kernel<<<SIM_BLOCKS, 32 * WAVES_PER_BLOCK, 0, stream>>>(
        emb, eidx, nidx, sims);
    hinge_kernel<<<dim3(POS_BLOCKS, NEG_SLICES), HINGE_THREADS, 0, stream>>>(
        pos, neg, partials);
    finalize_kernel<<<1, N_PARTIALS, 0, stream>>>(partials, (float*)d_out);
}
